// CommonNeighborsPredictor_14843406975379
// MI455X (gfx1250) — matrix operand order repaired and checked by reading, not compiled, not run on hardware
//
#include <hip/hip_runtime.h>
#include <math.h>

// Problem constants (from reference)
#define NN   10000      // nodes
#define DD   256        // embedding dim
#define QQ   4096       // query edges
#define NPAD 10016      // K padding for 32-wide WMMA K tiles (313*32 = 10016)
#define KT32 313        // ceil(NN/32)
#define CT16 625        // NN/16 candidate tiles

typedef __attribute__((ext_vector_type(16))) _Float16 v16h;
typedef __attribute__((ext_vector_type(8)))  float    v8f;

union U16 { v16h v; int4 q[2]; };   // 32-byte WMMA A/B operand

// ---------------------------------------------------------------------------
// Kernel 0: pack emb^T as f16 [DD][NPAD], zero-padded in K so GEMM-1 B tiles
// are two 16-byte K chunks per lane (matches 16-bit WMMA B layout).
// ---------------------------------------------------------------------------
__global__ __launch_bounds__(256) void pack_embT(const float* __restrict__ emb,
                                                 _Float16* __restrict__ embT) {
    int idx = blockIdx.x * 256 + threadIdx.x;
    if (idx >= DD * NPAD) return;
    int d = idx / NPAD;
    int k = idx - d * NPAD;
    float v = (k < NN) ? emb[(size_t)k * DD + d] : 0.0f;
    embT[idx] = (_Float16)v;
}

// ---------------------------------------------------------------------------
// Kernel 1: x = emb + (adj @ emb)/deg, then row-normalize -> xn (f16).
// Block = 16 adj rows x full 256 dims, 8 waves (each wave: 2 16x16 C tiles).
// Double-buffered pipeline: GLOBAL_LOAD_ASYNC_TO_LDS_B64 for tile k+1 is in
// flight while tile k feeds the WMMAs; async loads complete in order, so
// s_wait_asynccnt 1 after issuing k+1 guarantees tile k has landed. Only the
// final (ragged) tile takes a guarded synchronous path. Degree row-sum is
// fused into wave 0's A-operand gather (lanes m and m+16 cover all 32 K).
// ---------------------------------------------------------------------------
__global__ __launch_bounds__(256) void gemm_norm(const float* __restrict__ adj,
                                                 const float* __restrict__ emb,
                                                 const _Float16* __restrict__ embT,
                                                 _Float16* __restrict__ xn) {
    __shared__ float sAf[2][16 * 32];   // double-buffered adj tile (raw f32)
    __shared__ float sDeg[16];
    __shared__ float sN2[16];
    __shared__ float sInv[16];

    const int t    = threadIdx.x;
    const int lane = t & 31;
    const int wave = t >> 5;
    const int row0 = blockIdx.x * 16;

    if (t < 16) { sDeg[t] = 1e-6f; sN2[t] = 0.0f; }
    __syncthreads();

    v8f c0 = {}; v8f c1 = {};

    const int mlo = (lane < 16) ? 0 : 8;    // C layout: m = p + mlo
    const int klA = (lane < 16) ? 0 : 8;    // A layout K base
    const int klB = (lane < 16) ? 0 : 8;    // B layout K base
    const int col = lane & 15;
    const int d0  = wave * 32 + col;        // my dim (tile 0); tile 1 = d0+16

    // staging geometry: thread t moves 8 bytes (2 floats) of a 16x32 tile
    const int arow = (2 * t) / 32;          // my adj row (0..15)
    const int acol = (2 * t) % 32;          // my 2 cols
    const float* gbase = adj + (size_t)row0 * NN;             // uniform (SGPR)
    const unsigned gOffBase = (unsigned)(arow * NN + acol) * 4u;
    const unsigned ldsBase  = (unsigned)(size_t)(void*)&sAf[0][0] + (unsigned)t * 8u;
    const unsigned bufStride = 16u * 32u * 4u;                // 2 KB per buffer

    float dpart = 0.0f;                     // wave0: degree partial

    // prologue: kick off tile 0 into buffer 0
    asm volatile("global_load_async_to_lds_b64 %0, %1, %2"
                 :: "v"(ldsBase), "v"(gOffBase), "s"(gbase) : "memory");

    for (int kt = 0; kt < KT32; ++kt) {
        const int kk = kt * 32;
        const int nb = (kt + 1) & 1;        // next buffer
        if (kt + 1 < KT32 - 1) {
            // ---- prefetch tile kt+1 asynchronously into the other buffer ----
            const unsigned go = gOffBase + (unsigned)(kk + 32) * 4u;
            const unsigned ld = ldsBase + (unsigned)nb * bufStride;
            asm volatile("global_load_async_to_lds_b64 %0, %1, %2"
                         :: "v"(ld), "v"(go), "s"(gbase) : "memory");
            asm volatile("s_wait_asynccnt 1" ::: "memory");   // tile kt done (in-order)
        } else {
            if (kt + 1 == KT32 - 1) {
                // ---- guarded tail tile (runs once), synchronous into buf nb ----
                const int k0 = kk + 32 + acol;
                float v0 = (k0     < NN) ? gbase[(size_t)arow * NN + k0    ] : 0.0f;
                float v1 = (k0 + 1 < NN) ? gbase[(size_t)arow * NN + k0 + 1] : 0.0f;
                sAf[nb][arow * 32 + acol    ] = v0;
                sAf[nb][arow * 32 + acol + 1] = v1;
            }
            asm volatile("s_wait_asynccnt 0" ::: "memory");   // drain pipeline
        }
        __syncthreads();   // everyone's tile-kt data now visible in LDS

        // ---- A operand: gather 16 f32 from LDS, pack to f16 (0/1 exact) ----
        U16 a;
        {
            const float* pa = &sAf[kt & 1][col * 32 + klA];
            float4 f0 = *(const float4*)(pa);          // K = klA   .. klA+3
            float4 f1 = *(const float4*)(pa + 4);      // K = klA+4 .. klA+7
            float4 f2 = *(const float4*)(pa + 16);     // K = klA+16.. klA+19
            float4 f3 = *(const float4*)(pa + 20);     // K = klA+20.. klA+23
            if (wave == 0) {
                dpart += (f0.x + f0.y + f0.z + f0.w) + (f1.x + f1.y + f1.z + f1.w)
                       + (f2.x + f2.y + f2.z + f2.w) + (f3.x + f3.y + f3.z + f3.w);
            }
            a.v = (v16h){(_Float16)f0.x, (_Float16)f0.y, (_Float16)f0.z, (_Float16)f0.w,
                         (_Float16)f1.x, (_Float16)f1.y, (_Float16)f1.z, (_Float16)f1.w,
                         (_Float16)f2.x, (_Float16)f2.y, (_Float16)f2.z, (_Float16)f2.w,
                         (_Float16)f3.x, (_Float16)f3.y, (_Float16)f3.z, (_Float16)f3.w};
        }
        // ---- B operands: two 16B chunks (K klB..+7, klB+16..+23) per lane from embT (L2-resident) ----
        U16 b0, b1;
        {
            const _Float16* pb0 = embT + (size_t)d0 * NPAD + kk + klB;
            b0.q[0] = *(const int4*)(pb0);
            b0.q[1] = *(const int4*)(pb0 + 16);
            const _Float16* pb1 = embT + (size_t)(d0 + 16) * NPAD + kk + klB;
            b1.q[0] = *(const int4*)(pb1);
            b1.q[1] = *(const int4*)(pb1 + 16);
        }
        c0 = __builtin_amdgcn_wmma_f32_16x16x32_f16(false, a.v, false, b0.v,
                                                    (short)0, c0, false, false);
        c1 = __builtin_amdgcn_wmma_f32_16x16x32_f16(false, a.v, false, b1.v,
                                                    (short)0, c1, false, false);
        __syncthreads();   // reads of buf[kt&1] done before it is re-targeted
    }

    if (wave == 0) atomicAdd(&sDeg[col], dpart);   // lanes m and m+16 -> row m
    __syncthreads();

    // ---- epilogue: y = emb + acc/deg ; accumulate row |y|^2 ----
    float y[16];
#pragma unroll
    for (int p = 0; p < 8; ++p) {
        const int m = mlo + p;
        const float dg = sDeg[m];
        const size_t rb = (size_t)(row0 + m) * DD;
        float y0 = emb[rb + d0]      + c0[p] / dg;
        float y1 = emb[rb + d0 + 16] + c1[p] / dg;
        y[p] = y0; y[8 + p] = y1;
        atomicAdd(&sN2[m], y0 * y0 + y1 * y1);
    }
    __syncthreads();
    if (t < 16) sInv[t] = 1.0f / fmaxf(sqrtf(sN2[t]), 1e-8f);
    __syncthreads();
#pragma unroll
    for (int p = 0; p < 8; ++p) {
        const int m = mlo + p;
        const float inv = sInv[m];
        const size_t rb = (size_t)(row0 + m) * DD;
        xn[rb + d0]      = (_Float16)(y[p] * inv);
        xn[rb + d0 + 16] = (_Float16)(y[8 + p] * inv);
    }
}

// ---------------------------------------------------------------------------
// Kernel 2: fused masked cosine product. Block = 16 edges, 8 waves split the
// 625 candidate tiles. A_L/A_R (src/dst rows of xn) stay register-resident;
// one B tile feeds both WMMAs; mask applied on the f32 accumulator.
// ---------------------------------------------------------------------------
__global__ __launch_bounds__(256) void edge_score(const float* __restrict__ adj,
                                                  const _Float16* __restrict__ xn,
                                                  const int* __restrict__ edges,
                                                  float* __restrict__ out) {
    __shared__ int   sSrc[16], sDst[16];
    __shared__ float sAcc[16];

    const int t    = threadIdx.x;
    const int lane = t & 31;
    const int wave = t >> 5;
    const int eb   = blockIdx.x * 16;

    if (t < 16) {
        sSrc[t] = edges[eb + t];
        sDst[t] = edges[QQ + eb + t];
        sAcc[t] = 0.0f;
    }
    __syncthreads();

    const int mlo = (lane < 16) ? 0 : 8;
    const int klA = (lane < 16) ? 0 : 8;
    const int klB = (lane < 16) ? 0 : 8;
    const int col = lane & 15;

    // ---- A tiles: 16 edge rows of xn, 8 K-tiles each, per A layout ----
    U16 aL[8], aR[8];
    {
        const size_t rs = (size_t)sSrc[col] * DD;
        const size_t rd = (size_t)sDst[col] * DD;
#pragma unroll
        for (int kt = 0; kt < 8; ++kt) {
            const _Float16* p = xn + rs + kt * 32 + klA;
            aL[kt].q[0] = *(const int4*)(p);
            aL[kt].q[1] = *(const int4*)(p + 16);
            const _Float16* q = xn + rd + kt * 32 + klA;
            aR[kt].q[0] = *(const int4*)(q);
            aR[kt].q[1] = *(const int4*)(q + 16);
        }
    }
    // per-lane edge rows for the mask gathers (C layout: my m = mlo+p)
    int srcR[8], dstR[8];
#pragma unroll
    for (int p = 0; p < 8; ++p) { srcR[p] = sSrc[mlo + p]; dstR[p] = sDst[mlo + p]; }

    float wacc[8] = {};

    for (int ct = wave; ct < CT16; ct += 8) {
        const int cb = ct * 16;
        v8f cL = {}; v8f cR = {};
        const _Float16* pb = xn + (size_t)(cb + col) * DD + klB;
#pragma unroll
        for (int kt = 0; kt < 8; ++kt) {
            U16 b;
            b.q[0] = *(const int4*)(pb + kt * 32);
            b.q[1] = *(const int4*)(pb + kt * 32 + 16);
            cL = __builtin_amdgcn_wmma_f32_16x16x32_f16(false, aL[kt].v, false, b.v,
                                                        (short)0, cL, false, false);
            cR = __builtin_amdgcn_wmma_f32_16x16x32_f16(false, aR[kt].v, false, b.v,
                                                        (short)0, cR, false, false);
        }
        // mask = adj[src,c]*adj[dst,c] (exact 0/1) on C layout lanes
        const int c = cb + col;
        if (ct + 8 < CT16) {
            __builtin_prefetch(&adj[(size_t)srcR[0] * NN + c + 128], 0, 3);
            __builtin_prefetch(&adj[(size_t)dstR[0] * NN + c + 128], 0, 3);
        }
#pragma unroll
        for (int p = 0; p < 8; ++p) {
            const float mLv = adj[(size_t)srcR[p] * NN + c];
            const float mRv = adj[(size_t)dstR[p] * NN + c];
            wacc[p] += mLv * mRv * cL[p] * cR[p];
        }
    }

    // butterfly-reduce across the 16 lanes holding the same edge m
#pragma unroll
    for (int p = 0; p < 8; ++p) {
        float v = wacc[p];
        v += __shfl_xor(v, 1, 32);
        v += __shfl_xor(v, 2, 32);
        v += __shfl_xor(v, 4, 32);
        v += __shfl_xor(v, 8, 32);
        wacc[p] = v;
    }
    if (col == 0) {           // lanes 0 and 16
#pragma unroll
        for (int p = 0; p < 8; ++p) atomicAdd(&sAcc[mlo + p], wacc[p]);
    }
    __syncthreads();
    if (t < 16) out[eb + t] = 1.0f / (1.0f + expf(-sAcc[t]));
}

// ---------------------------------------------------------------------------
extern "C" void kernel_launch(void* const* d_in, const int* in_sizes, int n_in,
                              void* d_out, int out_size, void* d_ws, size_t ws_size,
                              hipStream_t stream) {
    const float* emb   = (const float*)d_in[0];     // [N, D] f32
    const float* adj   = (const float*)d_in[1];     // [N, N] f32 (0/1)
    const int*   edges = (const int*)d_in[2];       // [2, Q]
    float*       out   = (float*)d_out;             // [Q] f32

    // workspace: xn f16 [N][D] then embT f16 [D][NPAD]
    _Float16* xn   = (_Float16*)d_ws;
    _Float16* embT = (_Float16*)((char*)d_ws + (size_t)NN * DD * sizeof(_Float16));

    pack_embT<<<(DD * NPAD + 255) / 256, 256, 0, stream>>>(emb, embT);
    gemm_norm<<<NN / 16, 256, 0, stream>>>(adj, emb, embT, xn);
    edge_score<<<QQ / 16, 256, 0, stream>>>(adj, xn, edges, out);
}
